// MambaBlock_16355235463883
// MI455X (gfx1250) — matrix operand order repaired and checked by reading, not compiled, not run on hardware
//
#include <hip/hip_runtime.h>
#include <hip/hip_bf16.h>
#include <math.h>
#include <stdint.h>

typedef _Float16 v8h  __attribute__((ext_vector_type(8)));
typedef _Float16 v16h __attribute__((ext_vector_type(16)));
typedef float    v8f  __attribute__((ext_vector_type(8)));
typedef unsigned int u32x4 __attribute__((ext_vector_type(4)));
typedef int          i32x4 __attribute__((ext_vector_type(4)));
typedef int          i32x8 __attribute__((ext_vector_type(8)));

#define D_MODEL 1024
#define D_STATE 16
#define D_CONV  4
#define D_INNER 2048
#define DT_RANK 64
#define BATCH   4
#define LENGTH  2048
#define ROWS    (BATCH * LENGTH)   // 8192
#define NCHUNK  16
#define CHUNK   (LENGTH / NCHUNK)  // 128
#define NBC     (2 * D_STATE)      // 32 B/C floats per timestep

// ---------------------------------------------------------------------------
// Elementwise helpers
// ---------------------------------------------------------------------------
__global__ void k_cvt_f16(const float* __restrict__ in, _Float16* __restrict__ out, int n) {
  int i = blockIdx.x * blockDim.x + threadIdx.x;
  if (i < n) out[i] = (_Float16)in[i];
}

__global__ void k_negexp(const float* __restrict__ in, float* __restrict__ out, int n) {
  int i = blockIdx.x * blockDim.x + threadIdx.x;
  if (i < n) out[i] = -__expf(in[i]);
}

// ---------------------------------------------------------------------------
// WMMA GEMM:  C[M,N] (f32) = A[M,K] (f16, row-major, lda) * B[N,K]^T (f16, ldb)
// One wave per block; wave tile = (MT*16) x (NT*16).
// EPI: 0 = plain store, 1 = store + f16 copy of cols < auxN, 2 = bias+softplus.
// ---------------------------------------------------------------------------
template<int MT, int NT, int EPI>
__global__ __launch_bounds__(32)
void k_gemm(const _Float16* __restrict__ A, const _Float16* __restrict__ B,
            float* __restrict__ C, int K, int lda, int ldb, int ldc,
            const float* __restrict__ bias, _Float16* __restrict__ aux, int auxN)
{
  const int lane = threadIdx.x;
  const int lr = lane & 15;
  const int hs = lane >> 4;
  const int m0 = blockIdx.y * (MT * 16);
  const int n0 = blockIdx.x * (NT * 16);

  v8f zero = {};
  v8f acc[MT][NT];
#pragma unroll
  for (int mt = 0; mt < MT; ++mt)
#pragma unroll
    for (int nt = 0; nt < NT; ++nt) acc[mt][nt] = zero;

  const _Float16* ap0 = A + (size_t)(m0 + lr) * lda + hs * 16;
  const _Float16* bp0 = B + (size_t)(n0 + lr) * ldb + hs * 16;

  for (int k0 = 0; k0 < K; k0 += 32) {
    v16h af[MT], bf[NT];
#pragma unroll
    for (int mt = 0; mt < MT; ++mt) {
      const _Float16* ap = ap0 + (size_t)mt * 16 * lda + k0;
      v8h c0 = *(const v8h*)(ap);
      v8h c1 = *(const v8h*)(ap + 8);
      af[mt] = __builtin_shufflevector(c0, c1, 0, 1, 2, 3, 4, 5, 6, 7,
                                       8, 9, 10, 11, 12, 13, 14, 15);
    }
#pragma unroll
    for (int nt = 0; nt < NT; ++nt)
      bf[nt] = *(const v16h*)(bp0 + (size_t)nt * 16 * ldb + k0);

    if (k0 + 32 < K) {  // speculative prefetch of next K slab
      __builtin_prefetch(ap0 + k0 + 32, 0, 0);
      __builtin_prefetch(bp0 + k0 + 32, 0, 0);
    }

#pragma unroll
    for (int mt = 0; mt < MT; ++mt)
#pragma unroll
      for (int nt = 0; nt < NT; ++nt)
        acc[mt][nt] = __builtin_amdgcn_wmma_f32_16x16x32_f16(
            false, af[mt], false, bf[nt], (short)0, acc[mt][nt], false, false);
  }

#pragma unroll
  for (int mt = 0; mt < MT; ++mt)
#pragma unroll
    for (int nt = 0; nt < NT; ++nt) {
      const int col = n0 + nt * 16 + lr;
#pragma unroll
      for (int r = 0; r < 8; ++r) {
        const int row = m0 + mt * 16 + hs * 8 + r;
        float v = acc[mt][nt][r];
        if (EPI == 2) {  // bias + softplus
          v += bias[col];
          v = (v > 20.0f) ? v : __logf(1.0f + __expf(v));
        }
        C[(size_t)row * ldc + col] = v;
        if (EPI == 1) {  // extract dt columns as f16 for the next GEMM
          if (col < auxN) aux[(size_t)row * auxN + col] = (_Float16)v;
        }
      }
    }
}

// ---------------------------------------------------------------------------
// Causal depthwise conv (k=4, left pad 3) + bias + SiLU.
// ---------------------------------------------------------------------------
__global__ void k_conv_silu(const float* __restrict__ xz, const float* __restrict__ w,
                            const float* __restrict__ bconv, float* __restrict__ xc,
                            _Float16* __restrict__ xc16)
{
  int t = blockIdx.x * blockDim.x + threadIdx.x;
  if (t >= ROWS * D_INNER) return;
  const int d = t & (D_INNER - 1);
  const int row = t >> 11;
  const int l = row & (LENGTH - 1);
  float acc = bconv[d];
#pragma unroll
  for (int j = 0; j < D_CONV; ++j) {
    const int back = (D_CONV - 1) - j;
    if (l >= back)
      acc = fmaf(w[d * D_CONV + j], xz[(size_t)(row - back) * (2 * D_INNER) + d], acc);
  }
  const float s = acc * (1.0f / (1.0f + __expf(-acc)));
  xc[t] = s;
  xc16[t] = (_Float16)s;
}

// ---------------------------------------------------------------------------
// TDM: stage a CHUNK x 32 f32 B/C tile from dbc (row stride 96 elems, starting
// at column DT_RANK) into LDS. D# per CDNA5 ISA ch.8:
//   group0: [1:0]=count=1 | lds_addr | global_addr[56:0] | type=2
//   group1: data_size=4B(code 2), tensor_dim0=32, tensor_dim1=CHUNK,
//           tile_dim0=32, tile_dim1=CHUNK, tensor_dim0_stride=96
// ---------------------------------------------------------------------------
__device__ __forceinline__ void tdm_load_bc(const float* gptr, unsigned lds_off) {
  unsigned long long ga = (unsigned long long)(uintptr_t)gptr;
  u32x4 g0 = { 1u,                                      // count=1, user mode
               lds_off,                                 // lds_addr
               (unsigned)(ga & 0xffffffffull),          // global_addr[31:0]
               (unsigned)(((ga >> 32) & 0x01ffffffull) | 0x80000000u) }; // +type=2
  i32x8 g1 = { 0x00020000,                // data_size = 4B
               (int)((NBC & 0xffff) << 16),          // tensor_dim0 lo16 -> [63:48]
               (int)((CHUNK & 0xffff) << 16),        // tensor_dim1 lo16 -> [111:96]? ([95:80])
               (int)((NBC & 0xffff) << 16),          // tile_dim0 -> [127:112]
               (int)CHUNK,                           // tile_dim1 -> [143:128]
               96,                                   // tensor_dim0_stride lo32
               0, 0 };
  i32x4 gz = { 0, 0, 0, 0 };
#if defined(__clang_major__) && __clang_major__ >= 23
  i32x8 gz8 = { 0, 0, 0, 0, 0, 0, 0, 0 };
  __builtin_amdgcn_tensor_load_to_lds(g0, g1, gz, gz, gz8, 0);
#else
  __builtin_amdgcn_tensor_load_to_lds(g0, g1, gz, gz, 0);
#endif
}

// ---------------------------------------------------------------------------
// Chunked selective scan. h enters the recurrence linearly, so:
//  pass 1 (k_scan_chunk): per (b, chunk, d): local scan with h0=0 -> h_local,
//          and decay product P = prod exp(delta*A).     [parallel, 4096 waves]
//  pass 2 (k_scan_carry): per (b, d): 16-step combine H_in[c+1]=P*H_in+h_local.
//  pass 3 (k_scan_apply): per (b, chunk, d): re-scan from true H_in, emit
//          y = (C.h + D*x) * silu(z) as f16.            [parallel, 4096 waves]
// B/C tiles staged to LDS by the Tensor Data Mover.
// ---------------------------------------------------------------------------
__global__ __launch_bounds__(32)
void k_scan_chunk(const float* __restrict__ delta, const float* __restrict__ xc,
                  const float* __restrict__ dbc, const float* __restrict__ Aneg,
                  float* __restrict__ Pout, float* __restrict__ hlout)
{
  __shared__ float sBC[CHUNK * NBC];
  const int tid = threadIdx.x;
  const int d = blockIdx.x * 32 + tid;
  const int c = blockIdx.y;
  const int b = blockIdx.z;
  const int row0 = b * LENGTH + c * CHUNK;

  tdm_load_bc(dbc + (size_t)row0 * (DT_RANK + NBC) + DT_RANK,
              (unsigned)(uintptr_t)&sBC[0]);

  float h[D_STATE], p[D_STATE], a[D_STATE];
#pragma unroll
  for (int n = 0; n < D_STATE; ++n) {
    h[n] = 0.0f; p[n] = 1.0f;
    a[n] = Aneg[n * D_INNER + d];
  }
  __builtin_amdgcn_s_wait_tensorcnt(0);
  __syncthreads();

  for (int i = 0; i < CHUNK; ++i) {
    const int row = row0 + i;
    const float dlt = delta[(size_t)row * D_INNER + d];
    const float xv = xc[(size_t)row * D_INNER + d];
    const float dx = dlt * xv;
#pragma unroll
    for (int n = 0; n < D_STATE; ++n) {
      const float da = __expf(dlt * a[n]);
      h[n] = fmaf(da, h[n], sBC[i * NBC + n] * dx);
      p[n] *= da;
    }
  }
  const size_t base = ((size_t)(b * NCHUNK + c) * D_STATE) * D_INNER + d;
#pragma unroll
  for (int n = 0; n < D_STATE; ++n) {
    Pout[base + (size_t)n * D_INNER] = p[n];
    hlout[base + (size_t)n * D_INNER] = h[n];
  }
}

__global__ void k_scan_carry(const float* __restrict__ P, const float* __restrict__ hl,
                             float* __restrict__ Hin)
{
  int t = blockIdx.x * blockDim.x + threadIdx.x;
  if (t >= BATCH * D_INNER) return;
  const int d = t & (D_INNER - 1);
  const int b = t >> 11;
  float H[D_STATE];
#pragma unroll
  for (int n = 0; n < D_STATE; ++n) H[n] = 0.0f;
  for (int c = 0; c < NCHUNK; ++c) {
    const size_t base = ((size_t)(b * NCHUNK + c) * D_STATE) * D_INNER + d;
#pragma unroll
    for (int n = 0; n < D_STATE; ++n) {
      const size_t idx = base + (size_t)n * D_INNER;
      Hin[idx] = H[n];
      H[n] = fmaf(P[idx], H[n], hl[idx]);
    }
  }
}

__global__ __launch_bounds__(32)
void k_scan_apply(const float* __restrict__ delta, const float* __restrict__ xc,
                  const float* __restrict__ dbc, const float* __restrict__ xz,
                  const float* __restrict__ Aneg, const float* __restrict__ Dvec,
                  const float* __restrict__ Hin, _Float16* __restrict__ y16)
{
  __shared__ float sBC[CHUNK * NBC];
  const int tid = threadIdx.x;
  const int d = blockIdx.x * 32 + tid;
  const int c = blockIdx.y;
  const int b = blockIdx.z;
  const int row0 = b * LENGTH + c * CHUNK;

  tdm_load_bc(dbc + (size_t)row0 * (DT_RANK + NBC) + DT_RANK,
              (unsigned)(uintptr_t)&sBC[0]);

  const size_t base = ((size_t)(b * NCHUNK + c) * D_STATE) * D_INNER + d;
  float h[D_STATE], a[D_STATE];
#pragma unroll
  for (int n = 0; n < D_STATE; ++n) {
    h[n] = Hin[base + (size_t)n * D_INNER];
    a[n] = Aneg[n * D_INNER + d];
  }
  const float Dd = Dvec[d];
  __builtin_amdgcn_s_wait_tensorcnt(0);
  __syncthreads();

  for (int i = 0; i < CHUNK; ++i) {
    const int row = row0 + i;
    const float dlt = delta[(size_t)row * D_INNER + d];
    const float xv = xc[(size_t)row * D_INNER + d];
    const float dx = dlt * xv;
    float y = Dd * xv;
#pragma unroll
    for (int n = 0; n < D_STATE; ++n) {
      const float da = __expf(dlt * a[n]);
      h[n] = fmaf(da, h[n], sBC[i * NBC + n] * dx);
      y = fmaf(sBC[i * NBC + D_STATE + n], h[n], y);
    }
    const float zv = xz[(size_t)row * (2 * D_INNER) + D_INNER + d];
    const float sz = zv * (1.0f / (1.0f + __expf(-zv)));
    y16[(size_t)row * D_INNER + d] = (_Float16)(y * sz);
  }
}

// ---------------------------------------------------------------------------
// Launch
// ---------------------------------------------------------------------------
extern "C" void kernel_launch(void* const* d_in, const int* in_sizes, int n_in,
                              void* d_out, int out_size, void* d_ws, size_t ws_size,
                              hipStream_t stream) {
  (void)in_sizes; (void)n_in; (void)out_size; (void)ws_size;
  const float* x     = (const float*)d_in[0];
  const float* W_in  = (const float*)d_in[1];
  const float* convw = (const float*)d_in[2];
  const float* convb = (const float*)d_in[3];
  const float* W_x   = (const float*)d_in[4];
  const float* W_dt  = (const float*)d_in[5];
  const float* b_dt  = (const float*)d_in[6];
  const float* A_raw = (const float*)d_in[7];
  const float* Dv    = (const float*)d_in[8];
  const float* W_out = (const float*)d_in[9];
  float* out = (float*)d_out;

  char* ws = (char*)d_ws;
  size_t off = 0;
  auto carve = [&](size_t bytes) -> void* {
    void* p = ws + off;
    off += (bytes + 255) & ~(size_t)255;
    return p;
  };
  _Float16* x16    = (_Float16*)carve((size_t)ROWS * D_MODEL * 2);
  _Float16* Win16  = (_Float16*)carve((size_t)2 * D_INNER * D_MODEL * 2);
  _Float16* Wx16   = (_Float16*)carve((size_t)(DT_RANK + NBC) * D_INNER * 2);
  _Float16* Wdt16  = (_Float16*)carve((size_t)D_INNER * DT_RANK * 2);
  _Float16* Wout16 = (_Float16*)carve((size_t)D_MODEL * D_INNER * 2);
  float*    Aneg   = (float*)carve((size_t)D_STATE * D_INNER * 4);
  float*    xz     = (float*)carve((size_t)ROWS * 2 * D_INNER * 4);
  float*    xc     = (float*)carve((size_t)ROWS * D_INNER * 4);
  _Float16* xc16   = (_Float16*)carve((size_t)ROWS * D_INNER * 2);
  float*    dbc    = (float*)carve((size_t)ROWS * (DT_RANK + NBC) * 4);
  _Float16* dt16   = (_Float16*)carve((size_t)ROWS * DT_RANK * 2);
  float*    delta  = (float*)carve((size_t)ROWS * D_INNER * 4);
  _Float16* y16    = (_Float16*)carve((size_t)ROWS * D_INNER * 2);
  float*    Pscan  = (float*)carve((size_t)BATCH * NCHUNK * D_STATE * D_INNER * 4);
  float*    hloc   = (float*)carve((size_t)BATCH * NCHUNK * D_STATE * D_INNER * 4);
  float*    HinS   = (float*)carve((size_t)BATCH * NCHUNK * D_STATE * D_INNER * 4);

  const int CT = 256;
  auto cvt = [&](const float* src, _Float16* dst, int n) {
    k_cvt_f16<<<(n + CT - 1) / CT, CT, 0, stream>>>(src, dst, n);
  };
  cvt(x,     x16,    ROWS * D_MODEL);
  cvt(W_in,  Win16,  2 * D_INNER * D_MODEL);
  cvt(W_x,   Wx16,   (DT_RANK + NBC) * D_INNER);
  cvt(W_dt,  Wdt16,  D_INNER * DT_RANK);
  cvt(W_out, Wout16, D_MODEL * D_INNER);
  k_negexp<<<(D_STATE * D_INNER + CT - 1) / CT, CT, 0, stream>>>(A_raw, Aneg, D_STATE * D_INNER);

  // GEMM1: xz = x * W_in^T   (8192 x 4096, K=1024)
  k_gemm<2, 4, 0><<<dim3((2 * D_INNER) / 64, ROWS / 32), 32, 0, stream>>>(
      x16, Win16, xz, D_MODEL, D_MODEL, D_MODEL, 2 * D_INNER, nullptr, nullptr, 0);

  // Depthwise causal conv + SiLU
  k_conv_silu<<<(ROWS * D_INNER + CT - 1) / CT, CT, 0, stream>>>(xz, convw, convb, xc, xc16);

  // GEMM2: dbc = xc * W_x^T  (8192 x 96, K=2048), extract dt cols as f16
  k_gemm<2, 2, 1><<<dim3((DT_RANK + NBC) / 32, ROWS / 32), 32, 0, stream>>>(
      xc16, Wx16, dbc, D_INNER, D_INNER, D_INNER, DT_RANK + NBC,
      nullptr, dt16, DT_RANK);

  // GEMM3: delta = softplus(dt * W_dt^T + b_dt)  (8192 x 2048, K=64)
  k_gemm<2, 4, 2><<<dim3(D_INNER / 64, ROWS / 32), 32, 0, stream>>>(
      dt16, Wdt16, delta, DT_RANK, DT_RANK, DT_RANK, D_INNER, b_dt, nullptr, 0);

  // Chunked selective scan (TDM-staged B/C, 3 passes)
  dim3 sgrid(D_INNER / 32, NCHUNK, BATCH);
  k_scan_chunk<<<sgrid, 32, 0, stream>>>(delta, xc, dbc, Aneg, Pscan, hloc);
  k_scan_carry<<<(BATCH * D_INNER + CT - 1) / CT, CT, 0, stream>>>(Pscan, hloc, HinS);
  k_scan_apply<<<sgrid, 32, 0, stream>>>(delta, xc, dbc, xz, Aneg, Dv, HinS, y16);

  // GEMM4: out = y * W_out^T  (8192 x 1024, K=2048)
  k_gemm<2, 4, 0><<<dim3(D_MODEL / 64, ROWS / 32), 32, 0, stream>>>(
      y16, Wout16, out, D_INNER, D_INNER, D_INNER, D_MODEL, nullptr, nullptr, 0);
}
